// BaseGenerator_30940944400779
// MI455X (gfx1250) — matrix operand order repaired and checked by reading, not compiled, not run on hardware
//
#include <hip/hip_runtime.h>
#include <hip/hip_bf16.h>
#include <math.h>

// ---------------- problem constants ----------------
#define Bz    64
#define Sq    200
#define Ed    512
#define Hh    8
#define Ll    6
#define FFd   2048
#define HDd   32
#define DHd   64
#define NAOB  5
#define NATOM 40
#define NBOND 6
#define NQ    50
#define Mrows (Bz * Sq)      // 12800 = 100*128
#define SPADW 224            // S padded to multiple of 32 for WMMA k-dim
#define NEGINF (-__builtin_inff())

typedef __attribute__((ext_vector_type(16))) _Float16 v16h;
typedef __attribute__((ext_vector_type(8)))  float    v8f;

// CDNA5 16-bit A-fragment (16x32 MxK): lanes 0-15 hold K {0..7,16..23},
// lanes 16-31 hold K {8..15,24..31} for the same rows (ISA 7.12.2).
__device__ __forceinline__ int a_kmap(int lane, int e) {
    return ((lane >= 16) ? 8 : 0) + e + ((e >= 8) ? 8 : 0);
}
// 16-bit B-fragment (32x16 KxN): lanes 0-15 hold K {0..7,16..23}, lanes 16-31 K {8..15,24..31}.
__device__ __forceinline__ int b_kmap(int lane, int e) {
    return ((lane >= 16) ? 8 : 0) + e + ((e >= 8) ? 8 : 0);
}
__device__ __forceinline__ float gelu_exact(float v) {
    return 0.5f * v * (1.0f + erff(v * 0.70710678118654752440f));
}

// ---------------- f32 -> f16 conversion (for weights) ----------------
__launch_bounds__(256)
__global__ void cvt_f16_kernel(const float* __restrict__ in, _Float16* __restrict__ out, int n)
{
    int idx = blockIdx.x * 256 + threadIdx.x;
    if (idx < n) out[idx] = (_Float16)in[idx];
}

// ---------------- embedding: writes f32 master + f16 shadow ----------------
__launch_bounds__(256)
__global__ void embed_kernel(const int* __restrict__ aob, const int* __restrict__ aid,
                             const int* __restrict__ bid,
                             const float* __restrict__ t_aob, const float* __restrict__ t_atom,
                             const float* __restrict__ t_bond,
                             float* __restrict__ x, _Float16* __restrict__ xh)
{
    size_t idx = (size_t)blockIdx.x * 256 + threadIdx.x;     // over Mrows*Ed
    int e = (int)(idx & (Ed - 1));
    int row = (int)(idx >> 9);
    float v = (t_aob[(size_t)aob[row] * Ed + e] + t_atom[(size_t)aid[row] * Ed + e] +
               t_bond[(size_t)bid[row] * Ed + e]) * 22.62741699796952f;   // sqrt(512)
    x[idx] = v;
    xh[idx] = (_Float16)v;
}

// ---------------- per-head additive bias with causal + key-pad masks baked in
__launch_bounds__(256)
__global__ void bias_kernel(const int* __restrict__ adj, const int* __restrict__ aqm,
                            const int* __restrict__ bqm, const int* __restrict__ aob,
                            const float* __restrict__ adj_e, const float* __restrict__ aq_e,
                            const float* __restrict__ bq_e, const float* __restrict__ dist_e,
                            float* __restrict__ bias)
{
    size_t idx = (size_t)blockIdx.x * 256 + threadIdx.x;     // over B*S*S
    if (idx >= (size_t)Bz * Sq * Sq) return;
    int j = (int)(idx % Sq);
    int i = (int)((idx / Sq) % Sq);
    int b = (int)(idx / ((size_t)Sq * Sq));
    bool masked = (j > i) || (aob[b * Sq + j] == 0 /*PAD*/);
    int av = adj[idx], aqv = aqm[idx], bqv = bqm[idx];
    int dv = (i > j) ? (i - j) : (j - i);
    #pragma unroll
    for (int h = 0; h < Hh; ++h) {
        float v = masked ? NEGINF
                         : adj_e[av * Hh + h] + aq_e[aqv * Hh + h] + bq_e[bqv * Hh + h] +
                           dist_e[dv * Hh + h];
        bias[(((size_t)(b * Hh + h) * Sq) + i) * Sq + j] = v;
    }
}

// ---------------- WMMA GEMM with async global->LDS DMA ----------------
// C[M,N] = act(A[M,K] @ W[N,K]^T + bv[N]); A,W are f16; output f32 (Cf) or f16 (Ch).
// Tile 128x128, K-step 32, double-buffered dynamic LDS loaded via
// GLOBAL_LOAD_ASYNC_TO_LDS_B128 (ASYNCcnt pipelined). Kernel uses ONLY dynamic
// LDS so the raw LDS offsets used by the async ops start at 0.
// LDS layout (f16 elems): buf b at b*10240; sA rows stride 40 (80B, 16B-aligned,
// 20-bank stride => conflict-free b128 fragment reads); sB at +5120.
#define GEMM_LDSBUF 10240
__device__ __forceinline__ void issue_tile_async(const _Float16* A, const _Float16* W,
                                                 int m0, int n0, int N, int K, int k0,
                                                 int bufbase, int tid)
{
    unsigned long long abase = (unsigned long long)(uintptr_t)A;
    unsigned long long wbase = (unsigned long long)(uintptr_t)W;
    #pragma unroll
    for (int i = 0; i < 2; ++i) {                 // A tile: 128 rows x 32 f16
        int idx = tid + 256 * i;
        int row = idx >> 2, seg = idx & 3;        // 4 x 16B segments per row
        unsigned voff = (unsigned)((((m0 + row) * K) + k0 + seg * 8) * 2);
        unsigned lds  = (unsigned)((bufbase + row * 40 + seg * 8) * 2);
        asm volatile("global_load_async_to_lds_b128 %0, %1, %2"
                     :: "v"(lds), "v"(voff), "s"(abase) : "memory");
    }
    #pragma unroll
    for (int i = 0; i < 2; ++i) {                 // W tile: 128 rows x 32 f16
        int idx = tid + 256 * i;
        int row = idx >> 2, seg = idx & 3;
        int nrow = n0 + row;
        if (nrow >= N) nrow = N - 1;              // clamp (no fault); masked at store
        unsigned voff = (unsigned)(((nrow * K) + k0 + seg * 8) * 2);
        unsigned lds  = (unsigned)((bufbase + 5120 + row * 40 + seg * 8) * 2);
        asm volatile("global_load_async_to_lds_b128 %0, %1, %2"
                     :: "v"(lds), "v"(voff), "s"(wbase) : "memory");
    }
}

__launch_bounds__(256)
__global__ void gemm_wmma_kernel(const _Float16* __restrict__ A, const _Float16* __restrict__ W,
                                 const float* __restrict__ bv,
                                 float* __restrict__ Cf, _Float16* __restrict__ Ch,
                                 int M, int N, int K, int act_gelu)
{
    extern __shared__ _Float16 smem[];
    const int tid = threadIdx.x, lane = tid & 31, wave = tid >> 5;
    const int row16 = lane & 15;
    const int m0 = blockIdx.y * 128, n0 = blockIdx.x * 128;
    const int mg = wave & 3;        // rows  mg*32 .. mg*32+31 (two 16-row subtiles)
    const int ng = wave >> 2;       // cols  ng*64 .. ng*64+63 (four 16-col subtiles)
    v8f acc[2][4] = {};

    const int nk = K / 32;
    issue_tile_async(A, W, m0, n0, N, K, 0, 0, tid);
    for (int t = 0; t < nk; ++t) {
        const int cur = (t & 1) * GEMM_LDSBUF;
        if (t + 1 < nk) {
            issue_tile_async(A, W, m0, n0, N, K, (t + 1) * 32, ((t + 1) & 1) * GEMM_LDSBUF, tid);
            asm volatile("s_wait_asynccnt 0x4" ::: "memory");   // prev tile landed
        } else {
            asm volatile("s_wait_asynccnt 0x0" ::: "memory");
        }
        __syncthreads();
        v16h a0, a1, bf0, bf1, bf2, bf3;
        #pragma unroll
        for (int e = 0; e < 16; ++e) {
            int ak = a_kmap(lane, e), bk = b_kmap(lane, e);
            a0[e]  = smem[cur + (mg * 32 + row16) * 40 + ak];
            a1[e]  = smem[cur + (mg * 32 + 16 + row16) * 40 + ak];
            bf0[e] = smem[cur + 5120 + (ng * 64 + row16) * 40 + bk];
            bf1[e] = smem[cur + 5120 + (ng * 64 + 16 + row16) * 40 + bk];
            bf2[e] = smem[cur + 5120 + (ng * 64 + 32 + row16) * 40 + bk];
            bf3[e] = smem[cur + 5120 + (ng * 64 + 48 + row16) * 40 + bk];
        }
        acc[0][0] = __builtin_amdgcn_wmma_f32_16x16x32_f16(false, a0, false, bf0, (short)0, acc[0][0], false, false);
        acc[0][1] = __builtin_amdgcn_wmma_f32_16x16x32_f16(false, a0, false, bf1, (short)0, acc[0][1], false, false);
        acc[0][2] = __builtin_amdgcn_wmma_f32_16x16x32_f16(false, a0, false, bf2, (short)0, acc[0][2], false, false);
        acc[0][3] = __builtin_amdgcn_wmma_f32_16x16x32_f16(false, a0, false, bf3, (short)0, acc[0][3], false, false);
        acc[1][0] = __builtin_amdgcn_wmma_f32_16x16x32_f16(false, a1, false, bf0, (short)0, acc[1][0], false, false);
        acc[1][1] = __builtin_amdgcn_wmma_f32_16x16x32_f16(false, a1, false, bf1, (short)0, acc[1][1], false, false);
        acc[1][2] = __builtin_amdgcn_wmma_f32_16x16x32_f16(false, a1, false, bf2, (short)0, acc[1][2], false, false);
        acc[1][3] = __builtin_amdgcn_wmma_f32_16x16x32_f16(false, a1, false, bf3, (short)0, acc[1][3], false, false);
        __syncthreads();
    }

    // C/D layout: VGPR e -> M = e (+8 for lanes 16-31), N = lane&15
    const int radd = (lane >= 16) ? 8 : 0;
    #pragma unroll
    for (int mi2 = 0; mi2 < 2; ++mi2) {
        #pragma unroll
        for (int nj = 0; nj < 4; ++nj) {
            int col = n0 + ng * 64 + nj * 16 + row16;
            if (col < N) {
                float bias = bv[col];
                int rbase = m0 + mg * 32 + mi2 * 16 + radd;
                #pragma unroll
                for (int e = 0; e < 8; ++e) {
                    float v = acc[mi2][nj][e] + bias;
                    if (act_gelu) v = gelu_exact(v);
                    size_t o = (size_t)(rbase + e) * N + col;
                    if (Ch) Ch[o] = (_Float16)v;
                    else    Cf[o] = v;
                }
            }
        }
    }
}

// ---------------- attention: one (b, h, 32-row query tile) per block ----------------
__launch_bounds__(256)
__global__ void attn_kernel(const _Float16* __restrict__ qkv, const float* __restrict__ bias,
                            _Float16* __restrict__ attn_out)
{
    __shared__ _Float16 s_q[32][72];       // Q * 1/sqrt(DH)
    __shared__ _Float16 s_kv[64][72];      // K/V staging
    __shared__ float    s_s[32][SPADW];    // scores f32
    __shared__ _Float16 s_p[32][SPADW];    // probs f16
    const int b = blockIdx.z, h = blockIdx.y, q0 = blockIdx.x * 32;
    const int tid = threadIdx.x, lane = tid & 31, wave = tid >> 5;
    const int row16 = lane & 15;

    {   // load Q tile (fold in softmax scale)
        int r = tid >> 3, d = (tid & 7) * 8;
        int s = q0 + r;
        if (s < Sq) {
            const _Float16* qp = qkv + (size_t)(b * Sq + s) * 1536 + h * DHd + d;
            #pragma unroll
            for (int i = 0; i < 8; ++i) s_q[r][d + i] = (_Float16)((float)qp[i] * 0.125f);
        } else {
            #pragma unroll
            for (int i = 0; i < 8; ++i) s_q[r][d + i] = (_Float16)0.0f;
        }
    }
    __syncthreads();

    // phase 1: scores = (Q*scale) @ K^T, 4 chunks of 64 key rows
    const int mi = wave & 1;
    const int ji = wave >> 1;
    for (int jc = 0; jc < 4; ++jc) {
        int j0 = jc * 64;
        int r = tid >> 2, d = (tid & 3) * 16;
        int s = j0 + r;
        if (s < Sq) {
            const _Float16* kp = qkv + (size_t)(b * Sq + s) * 1536 + 512 + h * DHd + d;
            #pragma unroll
            for (int i = 0; i < 16; ++i) s_kv[r][d + i] = kp[i];
        } else {
            #pragma unroll
            for (int i = 0; i < 16; ++i) s_kv[r][d + i] = (_Float16)0.0f;
        }
        __syncthreads();
        if (j0 + ji * 16 < SPADW) {            // wave-uniform; EXEC stays full
            v8f c = {};
            #pragma unroll
            for (int ks = 0; ks < DHd; ks += 32) {
                v16h a, bb;
                #pragma unroll
                for (int e = 0; e < 16; ++e) {
                    a[e]  = s_q[mi * 16 + row16][ks + a_kmap(lane, e)];
                    bb[e] = s_kv[ji * 16 + row16][ks + b_kmap(lane, e)];
                }
                c = __builtin_amdgcn_wmma_f32_16x16x32_f16(false, a, false, bb, (short)0, c, false, false);
            }
            int rr = mi * 16 + ((lane >= 16) ? 8 : 0);
            int cc = j0 + ji * 16 + row16;
            #pragma unroll
            for (int e = 0; e < 8; ++e) s_s[rr + e][cc] = c[e];
        }
        __syncthreads();
    }

    {   // softmax (f32) with bias; 8 threads per row
        int r = tid >> 3, sub = tid & 7;
        int q = q0 + r;
        int qc = (q < Sq) ? q : 0;
        const float* bp = bias + (((size_t)(b * Hh + h) * Sq) + qc) * Sq;
        float vals[SPADW / 8];
        float mx = NEGINF;
        #pragma unroll
        for (int i = 0; i < SPADW / 8; ++i) {
            int j = sub * (SPADW / 8) + i;
            float v = (q < Sq && j < Sq) ? (s_s[r][j] + bp[j]) : NEGINF;
            vals[i] = v;
            mx = fmaxf(mx, v);
        }
        for (int o = 4; o >= 1; o >>= 1) mx = fmaxf(mx, __shfl_xor(mx, o, 8));
        float sum = 0.0f;
        #pragma unroll
        for (int i = 0; i < SPADW / 8; ++i) {
            float e = (vals[i] == NEGINF) ? 0.0f : expf(vals[i] - mx);
            vals[i] = e;
            sum += e;
        }
        for (int o = 4; o >= 1; o >>= 1) sum += __shfl_xor(sum, o, 8);
        float inv = (sum > 0.0f) ? (1.0f / sum) : 0.0f;
        #pragma unroll
        for (int i = 0; i < SPADW / 8; ++i)
            s_p[r][sub * (SPADW / 8) + i] = (_Float16)(vals[i] * inv);
    }
    __syncthreads();

    // phase 2: O = P @ V, 7 chunks of 32 value rows
    const int ni = wave >> 1;
    v8f oacc = {};
    for (int vc = 0; vc < SPADW / 32; ++vc) {
        int v0 = vc * 32;
        int r = tid >> 3, d = (tid & 7) * 8;
        int s = v0 + r;
        if (s < Sq) {
            const _Float16* vp = qkv + (size_t)(b * Sq + s) * 1536 + 1024 + h * DHd + d;
            #pragma unroll
            for (int i = 0; i < 8; ++i) s_kv[r][d + i] = vp[i];
        } else {
            #pragma unroll
            for (int i = 0; i < 8; ++i) s_kv[r][d + i] = (_Float16)0.0f;
        }
        __syncthreads();
        v16h a, bb;
        #pragma unroll
        for (int e = 0; e < 16; ++e) {
            a[e]  = s_p[mi * 16 + row16][v0 + a_kmap(lane, e)];
            bb[e] = s_kv[b_kmap(lane, e)][ni * 16 + row16];
        }
        oacc = __builtin_amdgcn_wmma_f32_16x16x32_f16(false, a, false, bb, (short)0, oacc, false, false);
        __syncthreads();
    }
    int rr = q0 + mi * 16 + ((lane >= 16) ? 8 : 0);
    int dd = ni * 16 + row16;
    #pragma unroll
    for (int e = 0; e < 8; ++e) {
        int s = rr + e;
        if (s < Sq) attn_out[(size_t)(b * Sq + s) * Ed + h * DHd + dd] = (_Float16)oacc[e];
    }
}

// ---------------- out = LayerNorm(xin + delta)*g + bt ; writes f32 + f16 shadow
__launch_bounds__(256)
__global__ void add_ln_kernel(const float* xin, const float* delta,
                              const float* __restrict__ g, const float* __restrict__ bt,
                              float* out, _Float16* __restrict__ outh)
{
    __shared__ float red[256];
    int row = blockIdx.x, t = threadIdx.x;
    size_t base = (size_t)row * Ed;
    float v0 = xin[base + t], v1 = xin[base + t + 256];
    if (delta) { v0 += delta[base + t]; v1 += delta[base + t + 256]; }
    red[t] = v0 + v1;
    __syncthreads();
    for (int o = 128; o > 0; o >>= 1) { if (t < o) red[t] += red[t + o]; __syncthreads(); }
    float mean = red[0] * (1.0f / Ed);
    __syncthreads();
    float d0 = v0 - mean, d1 = v1 - mean;
    red[t] = d0 * d0 + d1 * d1;
    __syncthreads();
    for (int o = 128; o > 0; o >>= 1) { if (t < o) red[t] += red[t + o]; __syncthreads(); }
    float inv = rsqrtf(red[0] * (1.0f / Ed) + 1e-5f);
    float y0 = d0 * inv * g[t] + bt[t];
    float y1 = d1 * inv * g[t + 256] + bt[t + 256];
    out[base + t] = y0;
    out[base + t + 256] = y1;
    outh[base + t] = (_Float16)y0;
    outh[base + t + 256] = (_Float16)y1;
}

// ---------------- aob logit masking ----------------
__launch_bounds__(256)
__global__ void aob_fixup_kernel(float* __restrict__ out)
{
    int row = blockIdx.x * 256 + threadIdx.x;
    if (row >= Mrows) return;
    float* p = out + (size_t)row * NAOB;
    p[0] = NEGINF;                       // PAD
    p[1] = NEGINF;                       // BOS
    if ((row % Sq) == 0) p[4] = NEGINF;  // BOND_TOK at position 0
}

// ---------------- edge logits: last-t-wins scatter over queue ids ----------------
__launch_bounds__(256)
__global__ void edge_kernel(const float* __restrict__ o0, const float* __restrict__ o1,
                            const int* __restrict__ aqm, float* __restrict__ edge)
{
    __shared__ int tstar[NQ];
    int row = blockIdx.x;                // b*S + s
    int b = row / Sq;
    int t = threadIdx.x;
    if (t < NQ) {
        int last = -1;
        const int* ap = aqm + (size_t)row * Sq;
        for (int j = 0; j < Sq; ++j)
            if (ap[j] == t) last = j;
        tstar[t] = last;
    }
    __syncthreads();
    for (int item = t; item < NQ * NBOND; item += 256) {
        int q = item / NBOND, v = item % NBOND;
        float r = NEGINF;
        if (q != 0) {
            int tt = tstar[q];
            if (tt >= 0) {
                const float* p0 = o0 + (size_t)row * (NBOND * HDd) + v * HDd;
                const float* p1 = o1 + (size_t)(b * Sq + tt) * (NBOND * HDd) + v * HDd;
                float acc = 0.0f;
                #pragma unroll
                for (int hh = 0; hh < HDd; ++hh) acc += p0[hh] * p1[hh];
                r = acc * 0.17677669529663687f;   // 32^-0.5
            }
        }
        edge[(size_t)row * (NQ * NBOND) + item] = r;
    }
}

// ---------------- host driver ----------------
// Input pointer order assumption (JAX pytree flattening of setup_inputs()):
// params leaves in sorted-key order (layers list of sorted dicts nested), then
// the 7 integer arrays in insertion order.
extern "C" void kernel_launch(void* const* d_in, const int* in_sizes, int n_in,
                              void* d_out, int out_size, void* d_ws, size_t ws_size,
                              hipStream_t stream)
{
    (void)in_sizes; (void)n_in; (void)out_size; (void)ws_size;
    auto F = [&](int i) { return (const float*)d_in[i]; };
    auto I = [&](int i) { return (const int*)d_in[i]; };

    const float* adj_emb    = F(0);
    const float* atomq_emb  = F(1);
    const float* bondq_emb  = F(2);
    const float* dist_emb   = F(3);
    const float* edge_b0    = F(4);
    const float* edge_b1    = F(5);
    const float* edge_w0    = F(6);
    const float* edge_w1    = F(7);
    const float* fin_b      = F(8);
    const float* fin_s      = F(9);
    const float* gen_aob_b  = F(10);
    const float* gen_aob_w  = F(11);
    const float* gen_atom_b = F(12);
    const float* gen_atom_w = F(13);
    // layers: leaves 14 + 12*l, sorted: b1,b2,in_b,in_w,ln1_b,ln1_s,ln2_b,ln2_s,out_b,out_w,w1,w2
    const float* tok_aob  = F(86);
    const float* tok_atom = F(87);
    const float* tok_bond = F(88);
    const int* aob = I(89);
    const int* aid = I(90);
    const int* bid = I(91);
    /* queueid_sequences = I(92) unused by the reference forward */
    const int* adj = I(93);
    const int* aqm = I(94);
    const int* bqm = I(95);

    char* wsb = (char*)d_ws;
    size_t off = 0;
    auto take = [&](size_t bytes) {
        char* p = wsb + off;
        off = (off + bytes + 255) & ~(size_t)255;
        return p;
    };
    float*    x     = (float*)take((size_t)Mrows * Ed * 4);
    _Float16* xh    = (_Float16*)take((size_t)Mrows * Ed * 2);
    _Float16* qkvh  = (_Float16*)take((size_t)Mrows * 3 * Ed * 2);
    _Float16* attnh = (_Float16*)take((size_t)Mrows * Ed * 2);
    float*    proj  = (float*)take((size_t)Mrows * Ed * 4);
    _Float16* midh  = (_Float16*)take((size_t)Mrows * FFd * 2);
    float*    biasb = (float*)take((size_t)Bz * Hh * Sq * Sq * 4);
    float*    o0    = (float*)take((size_t)Mrows * NBOND * HDd * 4);
    float*    o1    = (float*)take((size_t)Mrows * NBOND * HDd * 4);
    // f16 weight pool
    _Float16* w16[Ll][4];   // in_w, out_w, w1, w2
    for (int l = 0; l < Ll; ++l) {
        w16[l][0] = (_Float16*)take((size_t)3 * Ed * Ed * 2);
        w16[l][1] = (_Float16*)take((size_t)Ed * Ed * 2);
        w16[l][2] = (_Float16*)take((size_t)FFd * Ed * 2);
        w16[l][3] = (_Float16*)take((size_t)Ed * FFd * 2);
    }
    _Float16* aobw16  = (_Float16*)take((size_t)NAOB * Ed * 2);
    _Float16* atomw16 = (_Float16*)take((size_t)NATOM * Ed * 2);
    _Float16* e0w16   = (_Float16*)take((size_t)NBOND * HDd * Ed * 2);
    _Float16* e1w16   = (_Float16*)take((size_t)NBOND * HDd * Ed * 2);

    float* out_aob  = (float*)d_out;
    float* out_atom = out_aob + (size_t)Mrows * NAOB;
    float* out_edge = out_atom + (size_t)Mrows * NATOM;

    auto cvt = [&](const float* src, _Float16* dst, int n) {
        cvt_f16_kernel<<<(n + 255) / 256, 256, 0, stream>>>(src, dst, n);
    };
    // one-time weight conversion (negligible traffic vs compute)
    for (int l = 0; l < Ll; ++l) {
        int base = 14 + 12 * l;
        cvt(F(base + 3),  w16[l][0], 3 * Ed * Ed);
        cvt(F(base + 9),  w16[l][1], Ed * Ed);
        cvt(F(base + 10), w16[l][2], FFd * Ed);
        cvt(F(base + 11), w16[l][3], Ed * FFd);
    }
    cvt(gen_aob_w,  aobw16,  NAOB * Ed);
    cvt(gen_atom_w, atomw16, NATOM * Ed);
    cvt(edge_w0,    e0w16,   NBOND * HDd * Ed);
    cvt(edge_w1,    e1w16,   NBOND * HDd * Ed);

    embed_kernel<<<(Mrows * Ed) / 256, 256, 0, stream>>>(aob, aid, bid, tok_aob, tok_atom,
                                                         tok_bond, x, xh);
    bias_kernel<<<((size_t)Bz * Sq * Sq + 255) / 256, 256, 0, stream>>>(
        adj, aqm, bqm, aob, adj_emb, atomq_emb, bondq_emb, dist_emb, biasb);

    const size_t GLDS = 2 * GEMM_LDSBUF * sizeof(_Float16);   // 40960 B dynamic LDS
    for (int l = 0; l < Ll; ++l) {
        int base = 14 + 12 * l;
        const float* b1    = F(base + 0);
        const float* b2    = F(base + 1);
        const float* in_b  = F(base + 2);
        const float* ln1_b = F(base + 4);
        const float* ln1_s = F(base + 5);
        const float* ln2_b = F(base + 6);
        const float* ln2_s = F(base + 7);
        const float* out_b = F(base + 8);

        gemm_wmma_kernel<<<dim3((3 * Ed) / 128, Mrows / 128), 256, GLDS, stream>>>(
            xh, w16[l][0], in_b, nullptr, qkvh, Mrows, 3 * Ed, Ed, 0);
        attn_kernel<<<dim3((Sq + 31) / 32, Hh, Bz), 256, 0, stream>>>(qkvh, biasb, attnh);
        gemm_wmma_kernel<<<dim3(Ed / 128, Mrows / 128), 256, GLDS, stream>>>(
            attnh, w16[l][1], out_b, proj, nullptr, Mrows, Ed, Ed, 0);
        add_ln_kernel<<<Mrows, 256, 0, stream>>>(x, proj, ln1_s, ln1_b, x, xh);
        gemm_wmma_kernel<<<dim3(FFd / 128, Mrows / 128), 256, GLDS, stream>>>(
            xh, w16[l][2], b1, nullptr, midh, Mrows, FFd, Ed, 1 /*gelu*/);
        gemm_wmma_kernel<<<dim3(Ed / 128, Mrows / 128), 256, GLDS, stream>>>(
            midh, w16[l][3], b2, proj, nullptr, Mrows, Ed, FFd, 0);
        add_ln_kernel<<<Mrows, 256, 0, stream>>>(x, proj, ln2_s, ln2_b, x, xh);
    }
    add_ln_kernel<<<Mrows, 256, 0, stream>>>(x, nullptr, fin_s, fin_b, x, xh);

    gemm_wmma_kernel<<<dim3(1, Mrows / 128), 256, GLDS, stream>>>(
        xh, aobw16, gen_aob_b, out_aob, nullptr, Mrows, NAOB, Ed, 0);
    aob_fixup_kernel<<<(Mrows + 255) / 256, 256, 0, stream>>>(out_aob);
    gemm_wmma_kernel<<<dim3(1, Mrows / 128), 256, GLDS, stream>>>(
        xh, atomw16, gen_atom_b, out_atom, nullptr, Mrows, NATOM, Ed, 0);
    gemm_wmma_kernel<<<dim3((NBOND * HDd + 127) / 128, Mrows / 128), 256, GLDS, stream>>>(
        xh, e0w16, edge_b0, o0, nullptr, Mrows, NBOND * HDd, Ed, 0);
    gemm_wmma_kernel<<<dim3((NBOND * HDd + 127) / 128, Mrows / 128), 256, GLDS, stream>>>(
        xh, e1w16, edge_b1, o1, nullptr, Mrows, NBOND * HDd, Ed, 0);
    edge_kernel<<<Mrows, 256, 0, stream>>>(o0, o1, aqm, out_edge);
}
